// PCmer_21560735826484
// MI455X (gfx1250) — matrix operand order repaired and checked by reading, not compiled, not run on hardware
//
#include <hip/hip_runtime.h>
#include <hip/hip_bf16.h>
#include <math.h>

// ---------------------------------------------------------------------------
// CDNA5 WMMA types (wave32, v_wmma_f32_16x16x32_bf16)
// ---------------------------------------------------------------------------
typedef __attribute__((ext_vector_type(16))) __bf16 v16bf;
typedef __attribute__((ext_vector_type(8)))  __bf16 v8bf;
typedef __attribute__((ext_vector_type(8)))  float  v8f;
typedef __attribute__((ext_vector_type(4)))  int    v4i;

#define BM 128
#define BN 64
#define BK 32
#define BKP 40          // padded LDS row (80B) -> conflict-free b128 frag loads
#define NBF 266
#define FPAD 272        // 266 padded to 17*16; pad entries stored as 0

// gfx1250 async VMEM->LDS staging (compile-guarded; confirmed present R3)
#if defined(__has_builtin)
#if __has_builtin(__builtin_amdgcn_global_load_async_to_lds_b128) && \
    __has_builtin(__builtin_amdgcn_s_wait_asynccnt)
#define HAVE_ASYNC 1
#endif
#endif
#ifndef HAVE_ASYNC
#define HAVE_ASYNC 0
#endif

__device__ __forceinline__ void stage16B(const __bf16* src, __bf16* dst) {
#if HAVE_ASYNC
  __builtin_amdgcn_global_load_async_to_lds_b128(
      (__attribute__((address_space(1))) v4i*)src,
      (__attribute__((address_space(3))) v4i*)dst, 0, 0);
#else
  *(v8bf*)dst = *(const v8bf*)src;
#endif
}

__device__ __forceinline__ void stage_drain() {
#if HAVE_ASYNC
  __builtin_amdgcn_s_wait_asynccnt(0);
#endif
}

// A-fragment per ISA 7.12.2 (16-bit A 16x32): lane half h reads two 16B runs:
// k in [h*8, h*8+8) and [16+h*8, 16+h*8+8)
__device__ __forceinline__ v16bf load_fragA(const __bf16* row, int half) {
  v8bf lo = *(const v8bf*)(row + half * 8);
  v8bf hi = *(const v8bf*)(row + 16 + half * 8);
  v16bf r;
#pragma unroll
  for (int i = 0; i < 8; ++i) { r[i] = lo[i]; r[i + 8] = hi[i]; }
  return r;
}

// Stage one (BMxBK, BNxBK) tile pair into LDS buffers Al/Bl.
// fast: tile fully interior -> wide/async copies; else guarded scalar fill.
template <int TA, int TB>
__device__ __forceinline__ void stage_tile(
    __bf16* Al, __bf16* Bl, const __bf16* __restrict__ Ag,
    const __bf16* __restrict__ Bg, int tid, int m0, int n0, int kk, int M,
    int N, int K, long long lda, long long ldb, bool fast) {
  if (fast) {
    if (TA) {
      // A[k][m]: coalesced along m, scattered b16 LDS stores
#pragma unroll
      for (int i = 0; i < 16; ++i) {
        int idx = i * 256 + tid;
        int am = idx & 127, ak = idx >> 7;
        Al[am * BKP + ak] = Ag[(long long)(kk + ak) * lda + (m0 + am)];
      }
    } else {
      // A[m][k]: async 16B copies
#pragma unroll
      for (int p = 0; p < 2; ++p) {
        int idx = p * 256 + tid;
        int am = idx >> 2;
        int ak = (idx & 3) * 8;
        stage16B(Ag + (long long)(m0 + am) * lda + (kk + ak),
                 &Al[am * BKP + ak]);
      }
    }
    if (TB) {
      // B[n][k]: async 16B copies
      int bn = tid >> 2;
      int bk = (tid & 3) * 8;
      stage16B(Bg + (long long)(n0 + bn) * ldb + (kk + bk),
               &Bl[bn * BKP + bk]);
    } else {
      // B[k][n]: gather 8 k-values per (lane, n), one 16B LDS store
      int bn = tid & 63;
      int kg = (tid >> 6) * 8;
      v8bf tmp;
#pragma unroll
      for (int j = 0; j < 8; ++j)
        tmp[j] = Bg[(long long)(kk + kg + j) * ldb + (n0 + bn)];
      *(v8bf*)&Bl[bn * BKP + kg] = tmp;
    }
  } else {
    // guarded edge staging (boundary blocks / partial K-step)
#pragma unroll
    for (int i = 0; i < 16; ++i) {
      int idx = i * 256 + tid;
      int am = TA ? (idx & 127) : (idx >> 5);
      int ak = TA ? (idx >> 7) : (idx & 31);
      long long m = m0 + am, k = kk + ak;
      __bf16 v = (__bf16)0.f;
      if (m < M && k < K) v = TA ? Ag[k * lda + m] : Ag[m * lda + k];
      Al[am * BKP + ak] = v;
    }
#pragma unroll
    for (int i = 0; i < 8; ++i) {
      int idx = i * 256 + tid;
      int bn = idx >> 5, bk = idx & 31;
      long long n = n0 + bn, k = kk + bk;
      __bf16 v = (__bf16)0.f;
      if (n < N && k < K) v = TB ? Bg[n * ldb + k] : Bg[k * ldb + n];
      Bl[bn * BKP + bk] = v;
    }
  }
}

// ---------------------------------------------------------------------------
// Batched bf16 GEMM:  C[z] (+)= rowScale ⊙ (A[z] @ B[z] + bias)
//   TA: A stored [K,M];  TB: B stored [N,K].  Double-buffered async staging.
// Block: 256 thr = 8 waves (4x2); block tile 128x64; wave tile 32x32 (2x2 WMMA)
// ---------------------------------------------------------------------------
template <int TA, int TB>
__global__ __launch_bounds__(256)
void gemm_wmma(const __bf16* __restrict__ Ag0, const __bf16* __restrict__ Bg0,
               const float* __restrict__ bias, const float* __restrict__ rowScale,
               void* __restrict__ Cv,
               int M, int N, int K,
               long long lda, long long ldb, long long ldc,
               int addC, int outBf16, int zdiv,
               long long sAo, long long sAi, long long sBo, long long sBi,
               long long sCo, long long sCi, long long sRo, long long sRi) {
  __shared__ __align__(16) __bf16 Alds[2][BM * BKP];
  __shared__ __align__(16) __bf16 Blds[2][BN * BKP];

  const int tid = threadIdx.x;
  const int lane = tid & 31;
  const int wave = tid >> 5;
  const int wm = (wave & 3) * 32;
  const int wn = (wave >> 2) * 32;
  const int m0 = blockIdx.y * BM;
  const int n0 = blockIdx.x * BN;
  const int z = blockIdx.z;
  const long long zo = z / zdiv, zi = z % zdiv;

  const __bf16* Ag = Ag0 + zo * sAo + zi * sAi;
  const __bf16* Bg = Bg0 + zo * sBo + zi * sBi;

  const bool interior = (m0 + BM <= M) && (n0 + BN <= N);

  v8f acc[2][2];
#pragma unroll
  for (int i = 0; i < 2; ++i)
#pragma unroll
    for (int j = 0; j < 2; ++j)
#pragma unroll
      for (int e = 0; e < 8; ++e) acc[i][j][e] = 0.f;

  const int half = lane >> 4;
  const int mr = lane & 15;

  // prologue: stage first tile into buffer 0
  stage_tile<TA, TB>(Alds[0], Blds[0], Ag, Bg, tid, m0, n0, 0, M, N, K, lda,
                     ldb, interior && (BK <= K));
  int cur = 0;

  for (int kk = 0; kk < K; kk += BK) {
    stage_drain();      // staged data for `cur` has landed
    __syncthreads();    // ...and prior compute on `cur^1` is fully done
    const int nxt = kk + BK;
    if (nxt < K)
      stage_tile<TA, TB>(Alds[cur ^ 1], Blds[cur ^ 1], Ag, Bg, tid, m0, n0,
                         nxt, M, N, K, lda, ldb, interior && (nxt + BK <= K));

    const __bf16* Al = Alds[cur];
    const __bf16* Bl = Blds[cur];
    v16bf af[2], bfr[2];
#pragma unroll
    for (int s = 0; s < 2; ++s) {
      af[s] = load_fragA(&Al[(wm + s * 16 + mr) * BKP], half);
      // B frag (32x16): lane n = mr, k runs [h*8, h*8+8) and [16+h*8, 16+h*8+8)
      bfr[s] = load_fragA(&Bl[(wn + s * 16 + mr) * BKP], half);
    }
#pragma unroll
    for (int si = 0; si < 2; ++si)
#pragma unroll
      for (int sj = 0; sj < 2; ++sj)
        acc[si][sj] = __builtin_amdgcn_wmma_f32_16x16x32_bf16(
            false, af[si], false, bfr[sj], (short)0, acc[si][sj], false, false);
    cur ^= 1;
  }

  // Epilogue: bias, per-row scale (dinv), residual add, f32/bf16 store
  const float* rs = rowScale ? (rowScale + zo * sRo + zi * sRi) : nullptr;
  const long long coff = zo * sCo + zi * sCi;
  float* Cf = (float*)Cv + coff;
  __bf16* Ch = (__bf16*)Cv + coff;
  const int nc = lane & 15;
#pragma unroll
  for (int si = 0; si < 2; ++si)
#pragma unroll
    for (int sj = 0; sj < 2; ++sj)
#pragma unroll
      for (int r = 0; r < 8; ++r) {
        int gm = m0 + wm + si * 16 + half * 8 + r;
        int gn = n0 + wn + sj * 16 + nc;
        if (gm < M && gn < N) {
          float v = acc[si][sj][r];
          if (bias) v += bias[gn];
          if (rs) v *= rs[gm];
          long long o = (long long)gm * ldc + gn;
          if (outBf16) {
            Ch[o] = (__bf16)v;
          } else {
            if (addC) v += Cf[o];
            Cf[o] = v;
          }
        }
      }
}

// ---------------------------------------------------------------------------
// f32 -> bf16 conversion (plain; for [N,K]-stored weights)
// ---------------------------------------------------------------------------
__global__ __launch_bounds__(256)
void cvt_kernel(const float* __restrict__ in, __bf16* __restrict__ out,
                long long n) {
  long long i = (long long)blockIdx.x * 256 + threadIdx.x;
  if (i < n) out[i] = (__bf16)in[i];
}

// f32 [L][K][N] -> bf16 [L][N][K] (transpose so GEMM B takes the async TB=1
// path). blockIdx.y = layer.
__global__ __launch_bounds__(256)
void cvtT_kernel(const float* __restrict__ in, __bf16* __restrict__ out,
                 int K, int N) {
  long long l = blockIdx.y;
  long long i = (long long)blockIdx.x * 256 + threadIdx.x;
  long long KN = (long long)K * N;
  if (i < KN) {
    long long k = i / N, n = i % N;
    out[l * KN + n * K + k] = (__bf16)in[l * KN + i];
  }
}

// ---------------------------------------------------------------------------
// LayerNorm over D=512, one block per row; bf16 output (GEMM A operand)
// ---------------------------------------------------------------------------
__global__ __launch_bounds__(256)
void ln_kernel(const float* __restrict__ x, const float* __restrict__ w,
               const float* __restrict__ b, __bf16* __restrict__ out) {
  __shared__ float red[256];
  const int t = threadIdx.x;
  const long long row = blockIdx.x;
  const float* xr = x + row * 512;
  float v0 = xr[t], v1 = xr[t + 256];
  red[t] = v0 + v1;
  __syncthreads();
  for (int s = 128; s > 0; s >>= 1) {
    if (t < s) red[t] += red[t + s];
    __syncthreads();
  }
  float mean = red[0] * (1.f / 512.f);
  __syncthreads();
  float d0 = v0 - mean, d1 = v1 - mean;
  red[t] = d0 * d0 + d1 * d1;
  __syncthreads();
  for (int s = 128; s > 0; s >>= 1) {
    if (t < s) red[t] += red[t + s];
    __syncthreads();
  }
  float rstd = rsqrtf(red[0] * (1.f / 512.f) + 1e-5f);
  out[row * 512 + t] = (__bf16)(d0 * rstd * w[t] + b[t]);
  out[row * 512 + t + 256] = (__bf16)(d1 * rstd * w[t + 256] + b[t + 256]);
}

// ---------------------------------------------------------------------------
// Performer softmax-kernel feature map. grid (bh=32, ntile=128), 8 waves,
// wave handles 4 rows; proj cached in LDS; bf16 out [bh, n, FPAD] (pad = 0).
// ---------------------------------------------------------------------------
__global__ __launch_bounds__(256)
void feat_kernel(const float* __restrict__ qk, const float* __restrict__ proj,
                 __bf16* __restrict__ out, int isQuery) {
  __shared__ float projLds[NBF * 64];
  __shared__ float qrow[8 * 64];
  const int tid = threadIdx.x, lane = tid & 31, wave = tid >> 5;
  const int bh = blockIdx.x;
  const int b = bh >> 3, h = bh & 7;
  const int rowBase = blockIdx.y * 32;
  for (int i = tid; i < NBF * 64; i += 256) projLds[i] = proj[i];
  __syncthreads();
  const float normalizer = 0.35355339059327373f;  // 64^-0.25
  const float nrm2 = normalizer * normalizer;
  const float ratio = rsqrtf((float)NBF);
  for (int rb = 0; rb < 4; ++rb) {
    const int n = rowBase + rb * 8 + wave;
    const float* qptr = qk + ((long long)b * 4096 + n) * 512 + (long long)h * 64;
    float q0 = qptr[2 * lane], q1 = qptr[2 * lane + 1];
    qrow[wave * 64 + 2 * lane] = q0;
    qrow[wave * 64 + 2 * lane + 1] = q1;
    float ss = q0 * q0 + q1 * q1;
    for (int m = 16; m >= 1; m >>= 1) ss += __shfl_xor(ss, m, 32);
    const float diag = 0.5f * nrm2 * ss;
    float dd[9];
    float mx = -1e30f;
#pragma unroll
    for (int tch = 0; tch < 9; ++tch) {
      int j = lane + tch * 32;
      float a = 0.f;
      if (j < NBF) {
        const float* pr = &projLds[j * 64];
        const float* qr = &qrow[wave * 64];
#pragma unroll 8
        for (int d = 0; d < 64; ++d) a += qr[d] * pr[d];
        a *= normalizer;
        dd[tch] = a;
        mx = fmaxf(mx, a);
      } else {
        dd[tch] = 0.f;
      }
    }
    for (int m = 16; m >= 1; m >>= 1) mx = fmaxf(mx, __shfl_xor(mx, m, 32));
    __bf16* orow = out + ((long long)bh * 4096 + n) * FPAD;
#pragma unroll
    for (int tch = 0; tch < 9; ++tch) {
      int j = lane + tch * 32;
      if (j < FPAD) {
        float v = 0.f;
        if (j < NBF) {
          v = isQuery ? ratio * (expf(dd[tch] - diag - mx) + 1e-4f)
                      : ratio * expf(dd[tch] - diag + 1e-4f);  // eps inside exp
        }
        orow[j] = (__bf16)v;
      }
    }
    __syncthreads();
  }
}

// ksum[z][f] = sum_n kp[z][n][f]
__global__ __launch_bounds__(256)
void ksum_kernel(const __bf16* __restrict__ kp, float* __restrict__ ksum) {
  const int z = blockIdx.x;
  for (int f = threadIdx.x; f < FPAD; f += 256) {
    float s = 0.f;
    const __bf16* base = kp + (long long)z * 4096 * FPAD + f;
    for (int n = 0; n < 4096; ++n) s += (float)base[(long long)n * FPAD];
    ksum[z * FPAD + f] = s;
  }
}

// dinv[z][n] = 1/(qp[z][n] . ksum[z] + 1e-8)
__global__ __launch_bounds__(256)
void dinv_kernel(const __bf16* __restrict__ qp, const float* __restrict__ ksum,
                 float* __restrict__ dinv) {
  const int z = blockIdx.y;
  const int n = blockIdx.x * 256 + threadIdx.x;
  const __bf16* row = qp + ((long long)z * 4096 + n) * FPAD;
  float s = 0.f;
  for (int j = 0; j < FPAD; ++j) s += (float)row[j] * ksum[z * FPAD + j];
  dinv[(long long)z * 4096 + n] = 1.f / (s + 1e-8f);
}

// GLU: z[row][c] = t[row][c] * sigmoid(t[row][c+1024])
__global__ __launch_bounds__(256)
void glu_kernel(const float* __restrict__ t, float* __restrict__ zout) {
  long long i = (long long)blockIdx.x * 256 + threadIdx.x;  // 16384*1024
  long long row = i >> 10, c = i & 1023;
  float a = t[row * 2048 + c];
  float g = t[row * 2048 + 1024 + c];
  zout[i] = a * (1.f / (1.f + expf(-g)));
}

// depthwise conv (K=31, same pad) over n per batch, + bias + Swish, bf16 out
__global__ __launch_bounds__(256)
void dwconv_kernel(const float* __restrict__ zin, const float* __restrict__ w,
                   const float* __restrict__ bias, __bf16* __restrict__ out) {
  long long i = (long long)blockIdx.x * 256 + threadIdx.x;  // 16384*1024
  long long c = i & 1023;
  long long row = i >> 10;
  long long b = row >> 12, n = row & 4095;
  float acc = bias[c];
#pragma unroll
  for (int t = 0; t < 31; ++t) {
    long long nn = n + t - 15;
    if (nn >= 0 && nn < 4096)
      acc += zin[((b << 12) + nn) * 1024 + c] * w[c * 31 + t];
  }
  out[i] = (__bf16)(acc * (1.f / (1.f + expf(-acc))));
}

// ---------------------------------------------------------------------------
// Host driver
// ---------------------------------------------------------------------------
static void launch_gemm(int tA, int tB, const __bf16* A, long long lda,
                        const __bf16* B, long long ldb, const float* bias,
                        const float* rs, void* C, long long ldc, int outBf16,
                        int addC, int M, int N, int K, int nz, int zdiv,
                        long long sAo, long long sAi, long long sBo,
                        long long sBi, long long sCo, long long sCi,
                        long long sRo, long long sRi, hipStream_t stream) {
  dim3 grid((N + BN - 1) / BN, (M + BM - 1) / BM, nz);
  if (!tA && !tB)
    gemm_wmma<0, 0><<<grid, 256, 0, stream>>>(A, B, bias, rs, C, M, N, K, lda,
                                              ldb, ldc, addC, outBf16, zdiv,
                                              sAo, sAi, sBo, sBi, sCo, sCi,
                                              sRo, sRi);
  else if (!tA && tB)
    gemm_wmma<0, 1><<<grid, 256, 0, stream>>>(A, B, bias, rs, C, M, N, K, lda,
                                              ldb, ldc, addC, outBf16, zdiv,
                                              sAo, sAi, sBo, sBi, sCo, sCi,
                                              sRo, sRi);
  else
    gemm_wmma<1, 0><<<grid, 256, 0, stream>>>(A, B, bias, rs, C, M, N, K, lda,
                                              ldb, ldc, addC, outBf16, zdiv,
                                              sAo, sAi, sBo, sBi, sCo, sCi,
                                              sRo, sRi);
}

extern "C" void kernel_launch(void* const* d_in, const int* in_sizes, int n_in,
                              void* d_out, int out_size, void* d_ws,
                              size_t ws_size, hipStream_t stream) {
  (void)in_sizes; (void)n_in; (void)out_size; (void)ws_size;
  const float* x     = (const float*)d_in[0];
  const float* ln1_w = (const float*)d_in[1];
  const float* ln1_b = (const float*)d_in[2];
  const float* Wq    = (const float*)d_in[3];
  const float* bq    = (const float*)d_in[4];
  const float* Wk    = (const float*)d_in[5];
  const float* bk    = (const float*)d_in[6];
  const float* Wv    = (const float*)d_in[7];
  const float* bv    = (const float*)d_in[8];
  const float* Wo    = (const float*)d_in[9];
  const float* bo    = (const float*)d_in[10];
  const float* proj  = (const float*)d_in[11];
  const float* ln2_w = (const float*)d_in[12];
  const float* ln2_b = (const float*)d_in[13];
  const float* pw1_w = (const float*)d_in[14];
  const float* pw1_b = (const float*)d_in[15];
  const float* dw_w  = (const float*)d_in[16];
  const float* dw_b  = (const float*)d_in[17];
  const float* pw2_w = (const float*)d_in[18];
  const float* pw2_b = (const float*)d_in[19];

  const long long ROWS = 16384;  // b*n

  // ---- workspace layout (256B-aligned regions) ----
  char* base = (char*)d_ws;
  size_t off = 0;
  auto alloc = [&](size_t bytes) {
    size_t o = off;
    off += (bytes + 255) & ~(size_t)255;
    return base + o;
  };
  float*  xc    = (float*)alloc(ROWS * 512 * 4);
  __bf16* abuf  = (__bf16*)alloc(ROWS * 512 * 2);
  float*  qb    = (float*)alloc(ROWS * 512 * 4);
  float*  kb    = (float*)alloc(ROWS * 512 * 4);
  __bf16* vb    = (__bf16*)alloc(ROWS * 512 * 2);
  __bf16* qpb   = (__bf16*)alloc((size_t)32 * 4096 * FPAD * 2);
  __bf16* kpb   = (__bf16*)alloc((size_t)32 * 4096 * FPAD * 2);
  __bf16* ctxb  = (__bf16*)alloc((size_t)32 * FPAD * 64 * 2);
  float*  ksumb = (float*)alloc(32 * FPAD * 4);
  float*  dinvb = (float*)alloc((size_t)32 * 4096 * 4);
  float*  zb    = (float*)alloc(ROWS * 1024 * 4);
  __bf16* Wq_h  = (__bf16*)alloc((size_t)6 * 512 * 512 * 2);  // [N][K]
  __bf16* Wk_h  = (__bf16*)alloc((size_t)6 * 512 * 512 * 2);  // [N][K]
  __bf16* Wv_h  = (__bf16*)alloc((size_t)6 * 512 * 512 * 2);  // [N][K]
  __bf16* Wo_h  = (__bf16*)alloc((size_t)6 * 512 * 512 * 2);  // [N][K]
  __bf16* pw1_h = (__bf16*)alloc((size_t)6 * 2048 * 512 * 2); // already [N][K]
  __bf16* pw2_h = (__bf16*)alloc((size_t)6 * 512 * 1024 * 2); // already [N][K]
  // aliases: conv-phase buffers reuse the (dead) qp/kp region (142.6MB)
  float*  tb = (float*)qpb;   // pw1 out, 134.2MB, lives until GLU
  __bf16* ub = (__bf16*)qpb;  // conv out, 33.5MB, written after tb is dead
  __bf16* ob = (__bf16*)qb;   // attention out, q f32 buffer is dead by then

  // ---- weight conversion (once per call); QKV/Wo transposed to [N][K] ----
  dim3 gT((512 * 512 + 255) / 256, 6);
  cvtT_kernel<<<gT, 256, 0, stream>>>(Wq, Wq_h, 512, 512);
  cvtT_kernel<<<gT, 256, 0, stream>>>(Wk, Wk_h, 512, 512);
  cvtT_kernel<<<gT, 256, 0, stream>>>(Wv, Wv_h, 512, 512);
  cvtT_kernel<<<gT, 256, 0, stream>>>(Wo, Wo_h, 512, 512);
  const long long nP1 = 6LL * 2048 * 512;
  cvt_kernel<<<(nP1 + 255) / 256, 256, 0, stream>>>(pw1_w, pw1_h, nP1);
  const long long nP2 = 6LL * 512 * 1024;
  cvt_kernel<<<(nP2 + 255) / 256, 256, 0, stream>>>(pw2_w, pw2_h, nP2);

  (void)hipMemcpyAsync(xc, x, ROWS * 512 * 4, hipMemcpyDeviceToDevice, stream);

  const long long sA = (long long)4096 * FPAD;  // qp/kp per-(b,h) stride
  for (int l = 0; l < 6; ++l) {
    // --- Performer attention ---
    ln_kernel<<<ROWS, 256, 0, stream>>>(xc, ln1_w + l * 512, ln1_b + l * 512, abuf);
    launch_gemm(0, 1, abuf, 512, Wq_h + (long long)l * 512 * 512, 512,
                bq + l * 512, nullptr, qb, 512, 0, 0, ROWS, 512, 512, 1, 1,
                0, 0, 0, 0, 0, 0, 0, 0, stream);
    launch_gemm(0, 1, abuf, 512, Wk_h + (long long)l * 512 * 512, 512,
                bk + l * 512, nullptr, kb, 512, 0, 0, ROWS, 512, 512, 1, 1,
                0, 0, 0, 0, 0, 0, 0, 0, stream);
    launch_gemm(0, 1, abuf, 512, Wv_h + (long long)l * 512 * 512, 512,
                bv + l * 512, nullptr, vb, 512, 1, 0, ROWS, 512, 512, 1, 1,
                0, 0, 0, 0, 0, 0, 0, 0, stream);
    feat_kernel<<<dim3(32, 128), 256, 0, stream>>>(
        qb, proj + (long long)l * NBF * 64, qpb, 1);
    feat_kernel<<<dim3(32, 128), 256, 0, stream>>>(
        kb, proj + (long long)l * NBF * 64, kpb, 0);
    ksum_kernel<<<32, 256, 0, stream>>>(kpb, ksumb);
    dinv_kernel<<<dim3(16, 32), 256, 0, stream>>>(qpb, ksumb, dinvb);
    // ctx[z] = kp[z]^T @ v[z]   (M=FPAD, N=64, K=4096), bf16 out
    launch_gemm(1, 0, kpb, FPAD, vb, 512, nullptr, nullptr, ctxb, 64, 1, 0,
                FPAD, 64, 4096, 32, 8,
                8 * sA, sA, (long long)4096 * 512, 64,
                8LL * FPAD * 64, (long long)FPAD * 64, 0, 0, stream);
    // o[z] = dinv ⊙ (qp[z] @ ctx[z]) -> bf16 [b, n, h*64]
    launch_gemm(0, 0, qpb, FPAD, ctxb, 64, nullptr, dinvb, ob, 512, 1, 0,
                4096, 64, FPAD, 32, 8,
                8 * sA, sA, 8LL * FPAD * 64, (long long)FPAD * 64,
                (long long)4096 * 512, 64, 8LL * 4096, 4096, stream);
    // x += o @ Wo + bo
    launch_gemm(0, 1, ob, 512, Wo_h + (long long)l * 512 * 512, 512,
                bo + l * 512, nullptr, xc, 512, 0, 1, ROWS, 512, 512, 1, 1,
                0, 0, 0, 0, 0, 0, 0, 0, stream);

    // --- conformer conv module ---
    ln_kernel<<<ROWS, 256, 0, stream>>>(xc, ln2_w + l * 512, ln2_b + l * 512, abuf);
    launch_gemm(0, 1, abuf, 512, pw1_h + (long long)l * 2048 * 512, 512,
                pw1_b + l * 2048, nullptr, tb, 2048, 0, 0, ROWS, 2048, 512,
                1, 1, 0, 0, 0, 0, 0, 0, 0, 0, stream);
    glu_kernel<<<65536, 256, 0, stream>>>(tb, zb);
    dwconv_kernel<<<65536, 256, 0, stream>>>(
        zb, dw_w + (long long)l * 1024 * 31, dw_b + l * 1024, ub);
    launch_gemm(0, 1, ub, 1024, pw2_h + (long long)l * 512 * 1024, 1024,
                pw2_b + l * 512, nullptr, xc, 512, 0, 1, ROWS, 512, 1024,
                1, 1, 0, 0, 0, 0, 0, 0, 0, 0, stream);
  }

  (void)hipMemcpyAsync(d_out, xc, ROWS * 512 * 4, hipMemcpyDeviceToDevice, stream);
}
